// ImprovedGCN_85633057948049
// MI455X (gfx1250) — matrix operand order repaired and checked by reading, not compiled, not run on hardware
//
#include <hip/hip_runtime.h>
#include <hip/hip_bf16.h>
#include <math.h>

// ---------------------------------------------------------------- constants
#define NN   50000
#define EE   1000000
#define FF   128
#define HH   128
#define BB   64
#define CLS  10
#define HEADS 2
#define EPS  1e-5f
#define BS   256

// ---------------------------------------------------------------- types
typedef __attribute__((ext_vector_type(16))) __bf16         v16bf;
typedef __attribute__((ext_vector_type(8)))  float          v8f;
typedef __attribute__((ext_vector_type(8)))  unsigned short u16x8;

union Frag { v16bf bf; u16x8 us[2]; };

// round-to-nearest-even f32 -> bf16 bits
static __device__ __forceinline__ unsigned short f2bf(float f) {
    unsigned int u = __float_as_uint(f);
    u += 0x7FFFu + ((u >> 16) & 1u);
    return (unsigned short)(u >> 16);
}

static __device__ __forceinline__ float bnf(float v, float g, float b, float m, float var) {
    return (v - m) * rsqrtf(var + EPS) * g + b;
}

static __device__ __forceinline__ void atomicMaxF(float* addr, float val) {
    int old = __float_as_int(*addr);
    while (__int_as_float(old) < val) {
        int assumed = old;
        old = atomicCAS((int*)addr, assumed, __float_as_int(val));
        if (old == assumed) break;
    }
}

// ---------------------------------------------------------------- WMMA GEMM
// C[M x Ncols] = A[M x K](bf16, row-major) * Bt[Ncols x K](bf16, B transposed)
// one wave -> 16 rows x all columns; 8 waves per block.
__global__ void k_wmma_gemm(const unsigned short* __restrict__ A,
                            const unsigned short* __restrict__ Bt,
                            float* __restrict__ C,
                            int M, int K, int Ncols, int accumulate) {
    const int lane    = threadIdx.x & 31;
    const int wave    = threadIdx.x >> 5;
    const int rowTile = blockIdx.x * 8 + wave;
    if (rowTile * 16 >= M) return;                 // wave-uniform exit (EXEC stays all-ones)
    const int rowBase = rowTile * 16;
    const int r16  = lane & 15;
    const int hi   = lane >> 4;                    // 0 for lanes 0-15, 1 for 16-31
    const int aoff = hi * 8;                       // A frag: lanes 16-31 hold K+8..15 / K+24..31
    const int boff = hi * 8;                       // B frag: lanes 16-31 hold K+8..15 / K+24..31
    const int rbase = hi * 8;                      // C frag: lanes 16-31 hold M rows 8..15
    const size_t abase = (size_t)(rowBase + r16) * (size_t)K;

    for (int nt = 0; nt < (Ncols >> 4); ++nt) {
        const int col = (nt << 4) + r16;
        const size_t bbase = (size_t)col * (size_t)K;
        v8f acc = {};
        if (accumulate) {
            for (int r = 0; r < 8; ++r)
                acc[r] = C[(size_t)(rowBase + rbase + r) * Ncols + col];
        }
        for (int k0 = 0; k0 < K; k0 += 32) {
            Frag fa, fb;
            fa.us[0] = *reinterpret_cast<const u16x8*>(A  + abase + k0 + aoff);
            fa.us[1] = *reinterpret_cast<const u16x8*>(A  + abase + k0 + 16 + aoff);
            fb.us[0] = *reinterpret_cast<const u16x8*>(Bt + bbase + k0 + boff);
            fb.us[1] = *reinterpret_cast<const u16x8*>(Bt + bbase + k0 + 16 + boff);
            acc = __builtin_amdgcn_wmma_f32_16x16x32_bf16(
                      false, fa.bf, false, fb.bf, (short)0, acc, false, false);
        }
        for (int r = 0; r < 8; ++r)
            C[(size_t)(rowBase + rbase + r) * Ncols + col] = acc[r];
    }
}

// ---------------------------------------------------------------- small kernels
__global__ void k_f32_to_bf16(const float* __restrict__ in, unsigned short* __restrict__ out,
                              unsigned int n) {
    unsigned int t = blockIdx.x * blockDim.x + threadIdx.x;
    if (t < n) out[t] = f2bf(in[t]);
}

// W[K x Ncols] (row-major) -> Wt[Ncols x K] bf16
__global__ void k_wtrans_bf16(const float* __restrict__ W, unsigned short* __restrict__ Wt,
                              int K, int Ncols) {
    unsigned int t = blockIdx.x * blockDim.x + threadIdx.x;
    if (t >= (unsigned)(K * Ncols)) return;
    int k = t / Ncols, n = t % Ncols;
    Wt[(size_t)n * K + k] = f2bf(W[t]);
}

__global__ void k_fill(float* __restrict__ p, float v, unsigned int n) {
    unsigned int t = blockIdx.x * blockDim.x + threadIdx.x;
    if (t < n) p[t] = v;
}

__global__ void k_deg(const int* __restrict__ dst, const float* __restrict__ ew,
                      float* __restrict__ deg) {
    unsigned int e = blockIdx.x * blockDim.x + threadIdx.x;
    if (e < (unsigned)EE) atomicAdd(deg + dst[e], ew[e]);
}

__global__ void k_dis(float* __restrict__ deg_dis) {
    unsigned int i = blockIdx.x * blockDim.x + threadIdx.x;
    if (i >= (unsigned)NN) return;
    float d = deg_dis[i] + 1.0f;                   // +1 self-loop weight
    deg_dis[i] = (d > 0.0f) ? rsqrtf(fmaxf(d, 1e-12f)) : 0.0f;
}

// out1[d,:] += h[s,:] * (dis[s]*ew*dis[d]); 32 threads (x float4) per edge
__global__ void k_gcn_scatter(const int* __restrict__ src, const int* __restrict__ dst,
                              const float* __restrict__ ew, const float* __restrict__ dis,
                              const float* __restrict__ h, float* __restrict__ out) {
    unsigned int t = blockIdx.x * blockDim.x + threadIdx.x;
    unsigned int e = t >> 5; int g = (t & 31) << 2;
    if (e >= (unsigned)EE) return;
    int s = src[e], d = dst[e];
    float norm = dis[s] * ew[e] * dis[d];
    const float4 v = *reinterpret_cast<const float4*>(h + (size_t)s * HH + g);
    float* o = out + (size_t)d * HH + g;
    atomicAdd(o + 0, v.x * norm); atomicAdd(o + 1, v.y * norm);
    atomicAdd(o + 2, v.z * norm); atomicAdd(o + 3, v.w * norm);
}

// + self-loop term + bias, BN1, ReLU -> x1 (f32, in place) and x1 bf16
__global__ void k_gcn_epi(float* __restrict__ out1, const float* __restrict__ h,
                          const float* __restrict__ dis, const float* __restrict__ bias,
                          const float* __restrict__ g1, const float* __restrict__ b1,
                          const float* __restrict__ m1, const float* __restrict__ v1,
                          unsigned short* __restrict__ x1b) {
    unsigned int t = blockIdx.x * blockDim.x + threadIdx.x;
    if (t >= (unsigned)NN * HH) return;
    unsigned int i = t >> 7; int f = t & 127;
    float di = dis[i];
    float v = out1[t] + h[t] * di * di + bias[f];
    v = bnf(v, g1[f], b1[f], m1[f], v1[f]);
    v = fmaxf(v, 0.0f);
    out1[t] = v;
    x1b[t]  = f2bf(v);
}

__global__ void k_sage_scatter(const int* __restrict__ src, const int* __restrict__ dst,
                               const float* __restrict__ x1, float* __restrict__ agg,
                               float* __restrict__ cnt) {
    unsigned int t = blockIdx.x * blockDim.x + threadIdx.x;
    unsigned int e = t >> 5; int g = (t & 31) << 2;
    if (e >= (unsigned)EE) return;
    int s = src[e], d = dst[e];
    const float4 v = *reinterpret_cast<const float4*>(x1 + (size_t)s * HH + g);
    float* o = agg + (size_t)d * HH + g;
    atomicAdd(o + 0, v.x); atomicAdd(o + 1, v.y);
    atomicAdd(o + 2, v.z); atomicAdd(o + 3, v.w);
    if (g == 0) atomicAdd(cnt + d, 1.0f);
}

__global__ void k_mean_bf16(const float* __restrict__ agg, const float* __restrict__ cnt,
                            unsigned short* __restrict__ meanb) {
    unsigned int t = blockIdx.x * blockDim.x + threadIdx.x;
    if (t >= (unsigned)NN * HH) return;
    unsigned int i = t >> 7;
    meanb[t] = f2bf(agg[t] / fmaxf(cnt[i], 1.0f));
}

__global__ void k_sage_epi(const float* __restrict__ tbuf, const float* __restrict__ bias,
                           const float* __restrict__ g2, const float* __restrict__ b2,
                           const float* __restrict__ m2, const float* __restrict__ v2,
                           unsigned short* __restrict__ x2b) {
    unsigned int t = blockIdx.x * blockDim.x + threadIdx.x;
    if (t >= (unsigned)NN * HH) return;
    int f = t & 127;
    float v = tbuf[t] + bias[f];
    v = bnf(v, g2[f], b2[f], m2[f], v2[f]);
    x2b[t] = f2bf(fmaxf(v, 0.0f));
}

// attention scores per (node, head): dot over 128 features
__global__ void k_att(const float* __restrict__ xh, const float* __restrict__ att_s,
                      const float* __restrict__ att_d,
                      float* __restrict__ as_, float* __restrict__ ad_) {
    unsigned int t = blockIdx.x * blockDim.x + threadIdx.x;
    if (t >= (unsigned)NN * HEADS) return;
    unsigned int i = t >> 1; int hd = t & 1;
    const float* row = xh + (size_t)i * (HEADS * HH) + hd * HH;
    const float* ws = att_s + hd * HH;
    const float* wd = att_d + hd * HH;
    float ss = 0.0f, sd = 0.0f;
    for (int f = 0; f < HH; ++f) { float x = row[f]; ss += x * ws[f]; sd += x * wd[f]; }
    as_[t] = ss; ad_[t] = sd;
}

static __device__ __forceinline__ void edge_sd(unsigned int idx, const int* src, const int* dst,
                                               int& s, int& d) {
    if (idx < (unsigned)EE) { s = src[idx]; d = dst[idx]; }
    else { s = d = (int)(idx - EE); }              // self loops
}

__global__ void k_gat_max(const int* __restrict__ src, const int* __restrict__ dst,
                          const float* __restrict__ as_, const float* __restrict__ ad_,
                          float* __restrict__ m_) {
    unsigned int t = blockIdx.x * blockDim.x + threadIdx.x;
    if (t >= (unsigned)(EE + NN) * HEADS) return;
    unsigned int idx = t >> 1; int hd = t & 1;
    int s, d; edge_sd(idx, src, dst, s, d);
    float e = as_[s * 2 + hd] + ad_[d * 2 + hd];
    e = (e > 0.0f) ? e : 0.2f * e;                 // leaky_relu(0.2)
    atomicMaxF(m_ + d * 2 + hd, e);
}

__global__ void k_gat_exp(const int* __restrict__ src, const int* __restrict__ dst,
                          const float* __restrict__ as_, const float* __restrict__ ad_,
                          const float* __restrict__ m_, float* __restrict__ ex_,
                          float* __restrict__ z_) {
    unsigned int t = blockIdx.x * blockDim.x + threadIdx.x;
    if (t >= (unsigned)(EE + NN) * HEADS) return;
    unsigned int idx = t >> 1; int hd = t & 1;
    int s, d; edge_sd(idx, src, dst, s, d);
    float e = as_[s * 2 + hd] + ad_[d * 2 + hd];
    e = (e > 0.0f) ? e : 0.2f * e;
    float ev = __expf(e - m_[d * 2 + hd]);
    ex_[t] = ev;
    atomicAdd(z_ + d * 2 + hd, ev);
}

__global__ void k_gat_scatter(const int* __restrict__ src, const int* __restrict__ dst,
                              const float* __restrict__ xh, const float* __restrict__ ex_,
                              const float* __restrict__ z_, float* __restrict__ out3) {
    unsigned int t = blockIdx.x * blockDim.x + threadIdx.x;
    unsigned int idx = t >> 5; int g = (t & 31) << 2;
    if (idx >= (unsigned)(EE + NN)) return;
    int s, d; edge_sd(idx, src, dst, s, d);
    float a0 = ex_[idx * 2 + 0] / z_[d * 2 + 0];
    float a1 = ex_[idx * 2 + 1] / z_[d * 2 + 1];
    const float4 v0 = *reinterpret_cast<const float4*>(xh + (size_t)s * 256 + g);
    const float4 v1 = *reinterpret_cast<const float4*>(xh + (size_t)s * 256 + 128 + g);
    float* o = out3 + (size_t)d * HH + g;
    atomicAdd(o + 0, 0.5f * (v0.x * a0 + v1.x * a1));
    atomicAdd(o + 1, 0.5f * (v0.y * a0 + v1.y * a1));
    atomicAdd(o + 2, 0.5f * (v0.z * a0 + v1.z * a1));
    atomicAdd(o + 3, 0.5f * (v0.w * a0 + v1.w * a1));
}

__global__ void k_gat_epi(float* __restrict__ out3, const float* __restrict__ bias,
                          const float* __restrict__ g3, const float* __restrict__ b3,
                          const float* __restrict__ m3, const float* __restrict__ v3) {
    unsigned int t = blockIdx.x * blockDim.x + threadIdx.x;
    if (t >= (unsigned)NN * HH) return;
    int f = t & 127;
    float v = out3[t] + bias[f];
    v = bnf(v, g3[f], b3[f], m3[f], v3[f]);
    out3[t] = fmaxf(v, 0.0f);
}

// mean-pool of concat([x1, x3]) into pool[B x 256]
__global__ void k_pool(const int* __restrict__ batch, const float* __restrict__ x1,
                       const float* __restrict__ x3, float* __restrict__ pool,
                       float* __restrict__ bcnt) {
    unsigned int t = blockIdx.x * blockDim.x + threadIdx.x;
    unsigned int i = t >> 6; int g = (t & 63) << 2;
    if (i >= (unsigned)NN) return;
    int b = batch[i];
    const float4 v = (g < 128)
        ? *reinterpret_cast<const float4*>(x1 + (size_t)i * HH + g)
        : *reinterpret_cast<const float4*>(x3 + (size_t)i * HH + (g - 128));
    float* o = pool + (size_t)b * 256 + g;
    atomicAdd(o + 0, v.x); atomicAdd(o + 1, v.y);
    atomicAdd(o + 2, v.z); atomicAdd(o + 3, v.w);
    if (g == 0) atomicAdd(bcnt + b, 1.0f);
}

__global__ void k_mlp1(const float* __restrict__ pool, const float* __restrict__ bcnt,
                       const float* __restrict__ W, const float* __restrict__ bias,
                       const float* __restrict__ gm, const float* __restrict__ bm,
                       const float* __restrict__ mm, const float* __restrict__ vm,
                       float* __restrict__ hmid) {
    unsigned int t = blockIdx.x * blockDim.x + threadIdx.x;
    if (t >= (unsigned)BB * HH) return;
    unsigned int b = t >> 7; int j = t & 127;
    float inv = 1.0f / fmaxf(bcnt[b], 1.0f);
    float s = 0.0f;
    for (int k = 0; k < 2 * HH; ++k)
        s += pool[b * 256 + k] * inv * W[(size_t)k * HH + j];
    s += bias[j];
    s = bnf(s, gm[j], bm[j], mm[j], vm[j]);
    hmid[t] = fmaxf(s, 0.0f);
}

__global__ void k_out(const float* __restrict__ hmid, const float* __restrict__ W,
                      const float* __restrict__ bias, float* __restrict__ out) {
    unsigned int t = blockIdx.x * blockDim.x + threadIdx.x;
    if (t >= (unsigned)BB * CLS) return;
    unsigned int b = t / CLS; int c = t % CLS;
    float s = 0.0f;
    for (int k = 0; k < HH; ++k) s += hmid[b * HH + k] * W[(size_t)k * CLS + c];
    out[t] = s + bias[c];
}

// ---------------------------------------------------------------- launcher
static inline unsigned int gdiv(unsigned long long n, unsigned int b) {
    return (unsigned int)((n + b - 1) / b);
}

extern "C" void kernel_launch(void* const* d_in, const int* in_sizes, int n_in,
                              void* d_out, int out_size, void* d_ws, size_t ws_size,
                              hipStream_t stream) {
    // -------- inputs (setup_inputs order)
    const float* x       = (const float*)d_in[0];
    const int*   eidx    = (const int*)d_in[1];
    const float* ew      = (const float*)d_in[2];
    const int*   batch   = (const int*)d_in[3];
    const float* W_gcn   = (const float*)d_in[4];
    const float* b_gcn   = (const float*)d_in[5];
    const float *bn1_g = (const float*)d_in[6],  *bn1_b = (const float*)d_in[7],
                *bn1_m = (const float*)d_in[8],  *bn1_v = (const float*)d_in[9];
    const float* W_sl    = (const float*)d_in[10];
    const float* b_sl    = (const float*)d_in[11];
    const float* W_sr    = (const float*)d_in[12];
    const float *bn2_g = (const float*)d_in[13], *bn2_b = (const float*)d_in[14],
                *bn2_m = (const float*)d_in[15], *bn2_v = (const float*)d_in[16];
    const float* W_gat   = (const float*)d_in[17];
    const float* att_s   = (const float*)d_in[18];
    const float* att_d   = (const float*)d_in[19];
    const float* b_gat   = (const float*)d_in[20];
    const float *bn3_g = (const float*)d_in[21], *bn3_b = (const float*)d_in[22],
                *bn3_m = (const float*)d_in[23], *bn3_v = (const float*)d_in[24];
    const float* W_m1    = (const float*)d_in[25];
    const float* b_m1    = (const float*)d_in[26];
    const float *bnm_g = (const float*)d_in[27], *bnm_b = (const float*)d_in[28],
                *bnm_m = (const float*)d_in[29], *bnm_v = (const float*)d_in[30];
    const float* W_m2    = (const float*)d_in[31];
    const float* b_m2    = (const float*)d_in[32];
    const int* src = eidx;
    const int* dst = eidx + EE;
    float* out = (float*)d_out;

    // -------- workspace carve (256B aligned)
    size_t off = 0;
    auto carve = [&](size_t bytes) -> char* {
        char* p = (char*)d_ws + off;
        off += (bytes + 255) & ~(size_t)255;
        return p;
    };
    float*          buf_h   = (float*)carve((size_t)NN * HH * 4);          // h / agg / t / out3,x3
    float*          buf_x1  = (float*)carve((size_t)NN * HH * 4);          // out1 -> x1
    unsigned short* b16a    = (unsigned short*)carve((size_t)NN * HH * 2); // x_bf16 -> x1_bf16 -> x2_bf16
    unsigned short* b16b    = (unsigned short*)carve((size_t)NN * HH * 2); // mean_bf16
    float*          buf_xh  = (float*)carve((size_t)NN * 256 * 4);         // GAT per-head features
    float*          dis     = (float*)carve((size_t)NN * 4);
    float*          cnt     = (float*)carve((size_t)NN * 4);
    float*          as_     = (float*)carve((size_t)NN * 2 * 4);
    float*          ad_     = (float*)carve((size_t)NN * 2 * 4);
    float*          m_      = (float*)carve((size_t)NN * 2 * 4);
    float*          z_      = (float*)carve((size_t)NN * 2 * 4);
    float*          ex_     = (float*)carve((size_t)(EE + NN) * 2 * 4);
    float*          pool    = (float*)carve((size_t)BB * 256 * 4);
    float*          bcnt    = (float*)carve((size_t)BB * 4);
    float*          hmid    = (float*)carve((size_t)BB * HH * 4);
    unsigned short* Wgcn_t  = (unsigned short*)carve((size_t)HH * HH * 2);
    unsigned short* Wsl_t   = (unsigned short*)carve((size_t)HH * HH * 2);
    unsigned short* Wsr_t   = (unsigned short*)carve((size_t)HH * HH * 2);
    unsigned short* Wgat_t  = (unsigned short*)carve((size_t)HH * 256 * 2);
    (void)ws_size; (void)n_in; (void)in_sizes; (void)out_size;

    const unsigned long long NH  = (unsigned long long)NN * HH;
    const unsigned int gemmBlocks = gdiv(NN / 16, 8);   // 8 waves/block, 16 rows/wave

    // -------- weight prep (bf16, transposed)
    k_wtrans_bf16<<<gdiv(HH * HH, BS), BS, 0, stream>>>(W_gcn, Wgcn_t, HH, HH);
    k_wtrans_bf16<<<gdiv(HH * HH, BS), BS, 0, stream>>>(W_sl,  Wsl_t,  HH, HH);
    k_wtrans_bf16<<<gdiv(HH * HH, BS), BS, 0, stream>>>(W_sr,  Wsr_t,  HH, HH);
    k_wtrans_bf16<<<gdiv(HH * 256, BS), BS, 0, stream>>>(W_gat, Wgat_t, HH, 256);
    k_f32_to_bf16<<<gdiv(NH, BS), BS, 0, stream>>>(x, b16a, (unsigned int)NH);

    // -------- GCN
    hipMemsetAsync(dis, 0, (size_t)NN * 4, stream);
    k_deg<<<gdiv(EE, BS), BS, 0, stream>>>(dst, ew, dis);
    k_dis<<<gdiv(NN, BS), BS, 0, stream>>>(dis);
    k_wmma_gemm<<<gemmBlocks, BS, 0, stream>>>(b16a, Wgcn_t, buf_h, NN, HH, HH, 0);
    hipMemsetAsync(buf_x1, 0, (size_t)NN * HH * 4, stream);
    k_gcn_scatter<<<gdiv((unsigned long long)EE * 32, BS), BS, 0, stream>>>(src, dst, ew, dis, buf_h, buf_x1);
    k_gcn_epi<<<gdiv(NH, BS), BS, 0, stream>>>(buf_x1, buf_h, dis, b_gcn,
                                               bn1_g, bn1_b, bn1_m, bn1_v, b16a);

    // -------- SAGE
    hipMemsetAsync(buf_h, 0, (size_t)NN * HH * 4, stream);
    hipMemsetAsync(cnt, 0, (size_t)NN * 4, stream);
    k_sage_scatter<<<gdiv((unsigned long long)EE * 32, BS), BS, 0, stream>>>(src, dst, buf_x1, buf_h, cnt);
    k_mean_bf16<<<gdiv(NH, BS), BS, 0, stream>>>(buf_h, cnt, b16b);
    k_wmma_gemm<<<gemmBlocks, BS, 0, stream>>>(b16b, Wsl_t, buf_h, NN, HH, HH, 0);
    k_wmma_gemm<<<gemmBlocks, BS, 0, stream>>>(b16a, Wsr_t, buf_h, NN, HH, HH, 1);
    k_sage_epi<<<gdiv(NH, BS), BS, 0, stream>>>(buf_h, b_sl, bn2_g, bn2_b, bn2_m, bn2_v, b16a);

    // -------- GAT
    k_wmma_gemm<<<gemmBlocks, BS, 0, stream>>>(b16a, Wgat_t, buf_xh, NN, HH, 256, 0);
    k_att<<<gdiv((unsigned long long)NN * HEADS, BS), BS, 0, stream>>>(buf_xh, att_s, att_d, as_, ad_);
    k_fill<<<gdiv((unsigned long long)NN * 2, BS), BS, 0, stream>>>(m_, -1e30f, NN * 2);
    k_gat_max<<<gdiv((unsigned long long)(EE + NN) * 2, BS), BS, 0, stream>>>(src, dst, as_, ad_, m_);
    hipMemsetAsync(z_, 0, (size_t)NN * 2 * 4, stream);
    k_gat_exp<<<gdiv((unsigned long long)(EE + NN) * 2, BS), BS, 0, stream>>>(src, dst, as_, ad_, m_, ex_, z_);
    hipMemsetAsync(buf_h, 0, (size_t)NN * HH * 4, stream);
    k_gat_scatter<<<gdiv((unsigned long long)(EE + NN) * 32, BS), BS, 0, stream>>>(src, dst, buf_xh, ex_, z_, buf_h);
    k_gat_epi<<<gdiv(NH, BS), BS, 0, stream>>>(buf_h, b_gat, bn3_g, bn3_b, bn3_m, bn3_v);

    // -------- pool + MLP head
    hipMemsetAsync(pool, 0, (size_t)BB * 256 * 4, stream);
    hipMemsetAsync(bcnt, 0, (size_t)BB * 4, stream);
    k_pool<<<gdiv((unsigned long long)NN * 64, BS), BS, 0, stream>>>(batch, buf_x1, buf_h, pool, bcnt);
    k_mlp1<<<gdiv(BB * HH, BS), BS, 0, stream>>>(pool, bcnt, W_m1, b_m1,
                                                 bnm_g, bnm_b, bnm_m, bnm_v, hmid);
    k_out<<<gdiv(BB * CLS, BS), BS, 0, stream>>>(hmid, W_m2, b_m2, out);
}
